// BNNFC_57174604644867
// MI455X (gfx1250) — matrix operand order repaired and checked by reading, not compiled, not run on hardware
//
#include <hip/hip_runtime.h>
#include <hip/hip_bf16.h>

// ---- problem dims (GLIFR) ----
#define B_    64
#define T_    200
#define I_    512
#define H_    2048
#define O_    512
#define A_    2
#define DLY_  20
#define NCH_  (T_/DLY_)   // 10 chunks
#define DT_   0.05f
#define RMEM_ 0.1f

typedef __attribute__((ext_vector_type(16))) __bf16 v16bf;
typedef __attribute__((ext_vector_type(8)))  __bf16 v8bf;
typedef __attribute__((ext_vector_type(8)))  float  v8f;

// ---- GEMM tiling: 256 threads = 8 waves; block 128x128, K-step 32 ----
#define BM 128
#define BN 128
#define BK 32
#define LDA (BK+8)   // 40 elems = 80 B rows (16B-aligned)
#define LDB (BK+8)

static __device__ __forceinline__ v16bf cat8(v8bf lo, v8bf hi) {
  union { v16bf v; v8bf h[2]; } u;
  u.h[0] = lo; u.h[1] = hi;
  return u.v;
}

// CDNA5 async copy: global -> LDS, 16B per lane, tracked by ASYNCcnt.
// LDS operand is the low 32 bits of the generic pointer (flat->LDS aperture
// truncation, ISA 10.2); HW adds the wave's LDS_BASE.
static __device__ __forceinline__ void async_b128(const void* g, void* l) {
  asm volatile("global_load_async_to_lds_b128 %0, %1, off"
               :: "v"((unsigned int)(unsigned long long)l), "v"(g)
               : "memory");
}
static __device__ __forceinline__ void wait_async0() {
  asm volatile("s_wait_asynccnt 0" ::: "memory");
}

__global__ __launch_bounds__(256) void cvt_f32_bf16(const float* __restrict__ in,
                                                    __bf16* __restrict__ out, int n) {
  int i = blockIdx.x * 256 + threadIdx.x;
  if (i < n) out[i] = (__bf16)in[i];
}

// out[n][k] = (bf16) in[k][n]; K,N multiples of 32. Tiled for coalescing.
__global__ __launch_bounds__(256) void cvt_f32_bf16_T(const float* __restrict__ in,
                                                      __bf16* __restrict__ out,
                                                      int K, int N) {
  __shared__ float tile[32][33];
  int k0 = blockIdx.y * 32, n0 = blockIdx.x * 32;
  #pragma unroll
  for (int s = 0; s < 4; ++s) {
    int k = k0 + threadIdx.y + s * 8;
    tile[threadIdx.y + s * 8][threadIdx.x] = in[(long)k * N + n0 + threadIdx.x];
  }
  __syncthreads();
  #pragma unroll
  for (int s = 0; s < 4; ++s) {
    int n = n0 + threadIdx.y + s * 8;
    out[(long)n * K + k0 + threadIdx.x] = (__bf16)tile[threadIdx.x][threadIdx.y + s * 8];
  }
}

__global__ __launch_bounds__(256) void zero_f32(float* __restrict__ p, int n) {
  int i = blockIdx.x * 256 + threadIdx.x;
  if (i < n) p[i] = 0.f;
}

// syn[b*20+t, :] = x[b, 20c+t, :] @ Wiv  (+ F[b, 20(c-1)+t, :] @ Wlat if c>0)
// Weights arrive pre-transposed: wivT is (H x I), wlatT is (H x H), [n][k].
__global__ __launch_bounds__(256) void glifr_chunk_gemm(
    const __bf16* __restrict__ xbf,  const __bf16* __restrict__ wivT,
    const __bf16* __restrict__ F,    const __bf16* __restrict__ wlatT,
    float* __restrict__ syn, int c) {
  __shared__ __bf16 As[2][BM * LDA];
  __shared__ __bf16 Bs[2][BN * LDB];      // [n][k]

  const int KT1  = I_ / BK;               // 16
  const int KT   = KT1 + ((c > 0) ? H_ / BK : 0);
  const int bRow = blockIdx.y * BM;
  const int bCol = blockIdx.x * BN;

  const int lane = threadIdx.x & 31;
  const int wv   = threadIdx.x >> 5;
  const int wM   = wv & 3;                // 4 waves over M
  const int wN   = wv >> 2;               // 2 waves over N
  const int l15  = lane & 15;
  const int lhi  = lane >> 4;

  v8f acc[2][4];
  #pragma unroll
  for (int im = 0; im < 2; ++im)
    #pragma unroll
    for (int in = 0; in < 4; ++in)
      #pragma unroll
      for (int e = 0; e < 8; ++e) acc[im][in][e] = 0.f;

  auto load_tile = [&](int kt, int bi) {
    // A tile: 128 rows x 32 k, 16B async chunks, 2 per thread
    #pragma unroll
    for (int s = 0; s < 2; ++s) {
      int q  = threadIdx.x + s * 256;
      int rr = bRow + (q >> 2);
      int bb = rr / DLY_;
      int tt = rr - bb * DLY_;
      int kk = (q & 3) * 8;
      const __bf16* asrc = (kt < KT1)
          ? xbf + ((long)bb * T_ + c * DLY_ + tt) * I_ + kt * BK + kk
          : F   + ((long)bb * T_ + (c - 1) * DLY_ + tt) * H_ + (kt - KT1) * BK + kk;
      async_b128(asrc, &As[bi][(q >> 2) * LDA + kk]);
    }
    // B tile: 128 n-rows x 32 k from pre-transposed weights: contiguous both sides
    #pragma unroll
    for (int s = 0; s < 2; ++s) {
      int q  = threadIdx.x + s * 256;
      int n  = q >> 2;
      int k8 = (q & 3) * 8;
      const __bf16* bsrc = (kt < KT1)
          ? wivT  + (long)(bCol + n) * I_ + kt * BK + k8
          : wlatT + (long)(bCol + n) * H_ + (kt - KT1) * BK + k8;
      async_b128(bsrc, &Bs[bi][n * LDB + k8]);
    }
  };

  auto compute = [&](int bi) {
    v16bf af[2], bfv[4];
    #pragma unroll
    for (int im = 0; im < 2; ++im) {
      // ISA 16-bit A 16x32 layout: lane<16 -> K{0..7,16..23}, lane>=16 -> K{8..15,24..31}
      const __bf16* ar = &As[bi][(wM * 32 + im * 16 + l15) * LDA];
      af[im] = cat8(*(const v8bf*)(ar + lhi * 8), *(const v8bf*)(ar + 16 + lhi * 8));
    }
    #pragma unroll
    for (int in = 0; in < 4; ++in) {
      // ISA 16-bit B 32x16 layout: lane<16 -> K{0..7,16..23}, lane>=16 -> K{8..15,24..31}
      const __bf16* br = &Bs[bi][(wN * 64 + in * 16 + l15) * LDB];
      bfv[in] = cat8(*(const v8bf*)(br + lhi * 8), *(const v8bf*)(br + 16 + lhi * 8));
    }
    #pragma unroll
    for (int im = 0; im < 2; ++im)
      #pragma unroll
      for (int in = 0; in < 4; ++in)
        acc[im][in] = __builtin_amdgcn_wmma_f32_16x16x32_bf16(
            false, af[im], false, bfv[in], (short)0, acc[im][in], false, false);
  };

  load_tile(0, 0);
  wait_async0();
  __syncthreads();
  for (int kt = 0; kt < KT; ++kt) {
    int bi = kt & 1;
    if (kt + 1 < KT) load_tile(kt + 1, bi ^ 1);   // async copies overlap the WMMAs below
    compute(bi);
    wait_async0();
    __syncthreads();
  }

  #pragma unroll
  for (int im = 0; im < 2; ++im)
    #pragma unroll
    for (int in = 0; in < 4; ++in) {
      int row = bRow + wM * 32 + im * 16 + lhi * 8;
      int col = bCol + wN * 64 + in * 16 + l15;
      float* cp = syn + (long)row * H_ + col;
      #pragma unroll
      for (int e = 0; e < 8; ++e) cp[(long)e * H_] = acc[im][in][e];
    }
}

// Elementwise GLIFR recurrence over one 20-step chunk; state in registers.
__global__ __launch_bounds__(256) void glifr_scan(
    const float* __restrict__ syn,
    float* __restrict__ volt, float* __restrict__ firs, float* __restrict__ asc,
    const float* __restrict__ thresh, const float* __restrict__ k_m,
    const float* __restrict__ asc_amp, const float* __restrict__ asc_r,
    const float* __restrict__ asc_k,
    __bf16* __restrict__ F, int c) {
  int idx = blockIdx.x * 256 + threadIdx.x;   // b*H + h
  int b = idx >> 11;                          // /H_
  int h = idx & (H_ - 1);
  float v  = volt[idx], f = firs[idx];
  float a0 = asc[idx],  a1 = asc[B_ * H_ + idx];
  float amp0 = asc_amp[h], amp1 = asc_amp[H_ + h];
  float r0 = asc_r[h],     r1 = asc_r[H_ + h];
  float d0 = __expf(-DT_ * asc_k[h]), d1 = __expf(-DT_ * asc_k[H_ + h]);
  float th = thresh[h], km = k_m[h];
  #pragma unroll
  for (int t = 0; t < DLY_; ++t) {
    float s   = syn[((long)b * DLY_ + t) * H_ + h];
    float na0 = a0 * d0 + f * (r0 * a0 + amp0);
    float na1 = a1 * d1 + f * (r1 * a1 + amp1);
    a0 = na0; a1 = na1;
    float tot = s + a0 + a1;               // I0 = 0
    v = v * (1.f - f);                     // V_RESET = 0
    v = v + DT_ * km * (RMEM_ * tot - v);
    f = 1.f / (1.f + __expf(-(v - th)));   // SIGMA_V = 1
    F[((long)b * T_ + c * DLY_ + t) * H_ + h] = (__bf16)f;
  }
  volt[idx] = v; firs[idx] = f;
  asc[idx] = a0; asc[B_ * H_ + idx] = a1;
}

// out[b*T+t, :] = F[b*T+t, :] @ Wout + b_out      (woutT is O x H, [n][k])
__global__ __launch_bounds__(256) void glifr_out_gemm(
    const __bf16* __restrict__ F, const __bf16* __restrict__ woutT,
    const float* __restrict__ bias, float* __restrict__ out) {
  __shared__ __bf16 As[2][BM * LDA];
  __shared__ __bf16 Bs[2][BN * LDB];

  const int KT   = H_ / BK;                 // 64
  const int bRow = blockIdx.y * BM;
  const int bCol = blockIdx.x * BN;

  const int lane = threadIdx.x & 31;
  const int wv   = threadIdx.x >> 5;
  const int wM   = wv & 3;
  const int wN   = wv >> 2;
  const int l15  = lane & 15;
  const int lhi  = lane >> 4;

  v8f acc[2][4];
  #pragma unroll
  for (int im = 0; im < 2; ++im)
    #pragma unroll
    for (int in = 0; in < 4; ++in)
      #pragma unroll
      for (int e = 0; e < 8; ++e) acc[im][in][e] = 0.f;

  auto load_tile = [&](int kt, int bi) {
    #pragma unroll
    for (int s = 0; s < 2; ++s) {
      int q  = threadIdx.x + s * 256;
      int kk = (q & 3) * 8;
      async_b128(F + (long)(bRow + (q >> 2)) * H_ + kt * BK + kk,
                 &As[bi][(q >> 2) * LDA + kk]);
    }
    #pragma unroll
    for (int s = 0; s < 2; ++s) {
      int q  = threadIdx.x + s * 256;
      int n  = q >> 2;
      int k8 = (q & 3) * 8;
      async_b128(woutT + (long)(bCol + n) * H_ + kt * BK + k8,
                 &Bs[bi][n * LDB + k8]);
    }
  };

  auto compute = [&](int bi) {
    v16bf af[2], bfv[4];
    #pragma unroll
    for (int im = 0; im < 2; ++im) {
      const __bf16* ar = &As[bi][(wM * 32 + im * 16 + l15) * LDA];
      af[im] = cat8(*(const v8bf*)(ar + lhi * 8), *(const v8bf*)(ar + 16 + lhi * 8));
    }
    #pragma unroll
    for (int in = 0; in < 4; ++in) {
      const __bf16* br = &Bs[bi][(wN * 64 + in * 16 + l15) * LDB];
      bfv[in] = cat8(*(const v8bf*)(br + lhi * 8), *(const v8bf*)(br + 16 + lhi * 8));
    }
    #pragma unroll
    for (int im = 0; im < 2; ++im)
      #pragma unroll
      for (int in = 0; in < 4; ++in)
        acc[im][in] = __builtin_amdgcn_wmma_f32_16x16x32_bf16(
            false, af[im], false, bfv[in], (short)0, acc[im][in], false, false);
  };

  load_tile(0, 0);
  wait_async0();
  __syncthreads();
  for (int kt = 0; kt < KT; ++kt) {
    int bi = kt & 1;
    if (kt + 1 < KT) load_tile(kt + 1, bi ^ 1);
    compute(bi);
    wait_async0();
    __syncthreads();
  }

  #pragma unroll
  for (int im = 0; im < 2; ++im)
    #pragma unroll
    for (int in = 0; in < 4; ++in) {
      int row = bRow + wM * 32 + im * 16 + lhi * 8;
      int col = bCol + wN * 64 + in * 16 + l15;
      float bo = bias[col];
      float* cp = out + (long)row * O_ + col;
      #pragma unroll
      for (int e = 0; e < 8; ++e) cp[(long)e * O_] = acc[im][in][e] + bo;
    }
}

extern "C" void kernel_launch(void* const* d_in, const int* in_sizes, int n_in,
                              void* d_out, int out_size, void* d_ws, size_t ws_size,
                              hipStream_t stream) {
  (void)in_sizes; (void)n_in; (void)out_size; (void)ws_size;
  const float* x       = (const float*)d_in[0];
  const float* w_iv    = (const float*)d_in[1];
  const float* w_lat   = (const float*)d_in[2];
  const float* thresh  = (const float*)d_in[3];
  const float* k_m     = (const float*)d_in[4];
  const float* asc_amp = (const float*)d_in[5];
  const float* asc_r   = (const float*)d_in[6];
  const float* asc_k   = (const float*)d_in[7];
  const float* w_out   = (const float*)d_in[8];
  const float* b_out   = (const float*)d_in[9];

  char*  ws  = (char*)d_ws;
  size_t off = 0;
  auto carve = [&](size_t bytes) -> char* {
    char* p = ws + off;
    off += (bytes + 255) & ~(size_t)255;
    return p;
  };
  __bf16* xbf   = (__bf16*)carve((size_t)B_ * T_ * I_ * 2);   // 12.5 MB
  __bf16* wivT  = (__bf16*)carve((size_t)I_ * H_ * 2);        //  2  MB  (H x I)
  __bf16* wlatT = (__bf16*)carve((size_t)H_ * H_ * 2);        //  8  MB  (H x H, L2-resident)
  __bf16* woutT = (__bf16*)carve((size_t)H_ * O_ * 2);        //  2  MB  (O x H)
  __bf16* F     = (__bf16*)carve((size_t)B_ * T_ * H_ * 2);   // 50  MB firing_seq (bf16)
  float*  syn   = (float*) carve((size_t)B_ * DLY_ * H_ * 4); // 10  MB per-chunk drive
  float*  volt  = (float*) carve((size_t)B_ * H_ * 4);
  float*  firs  = (float*) carve((size_t)B_ * H_ * 4);
  float*  asc   = (float*) carve((size_t)A_ * B_ * H_ * 4);

  cvt_f32_bf16<<<dim3((B_ * T_ * I_ + 255) / 256), 256, 0, stream>>>(x, xbf, B_ * T_ * I_);
  // transposed bf16 weights: [n][k] so GEMM B tiles are contiguous 16B moves
  cvt_f32_bf16_T<<<dim3(H_ / 32, I_ / 32), dim3(32, 8), 0, stream>>>(w_iv,  wivT,  I_, H_);
  cvt_f32_bf16_T<<<dim3(H_ / 32, H_ / 32), dim3(32, 8), 0, stream>>>(w_lat, wlatT, H_, H_);
  cvt_f32_bf16_T<<<dim3(O_ / 32, H_ / 32), dim3(32, 8), 0, stream>>>(w_out, woutT, H_, O_);

  // volt/firs/asc are contiguous (all sizes 256B-aligned): zero all state each call
  zero_f32<<<dim3((4 * B_ * H_ + 255) / 256), 256, 0, stream>>>(volt, 4 * B_ * H_);

  for (int c = 0; c < NCH_; ++c) {
    glifr_chunk_gemm<<<dim3(H_ / BN, (B_ * DLY_) / BM), 256, 0, stream>>>(
        xbf, wivT, F, wlatT, syn, c);
    glifr_scan<<<dim3((B_ * H_) / 256), 256, 0, stream>>>(
        syn, volt, firs, asc, thresh, k_m, asc_amp, asc_r, asc_k, F, c);
  }
  glifr_out_gemm<<<dim3(O_ / BN, (B_ * T_) / BM), 256, 0, stream>>>(
      F, woutT, b_out, (float*)d_out);
}
